// System2Graph_62740882260558
// MI455X (gfx1250) — matrix operand order repaired and checked by reading, not compiled, not run on hardware
//
#include <hip/hip_runtime.h>
#include <hip/hip_bf16.h>
#include <math.h>

// ---- problem constants (from reference) ----
#define D_DIM 1024
#define NF_FN 4
#define NTOK  4096          // B*T
#define NROWS 16384         // NTOK*K
#define TD    3072          // 3*D
#define TD2   1536          // TD/2 (bf16 pairs per row)
#define NKT   96            // TD/32 K-tiles
#define NDT   64            // D/16 column tiles
#define NSTAGE 12           // TD/256 K-stages
#define MAXD_ 256
#define NC_   32
#define LN_EPS 1e-5f

typedef __attribute__((ext_vector_type(16))) __bf16 v16bf;
typedef __attribute__((ext_vector_type(8)))  float  v8f;

union AFrag { v16bf v; unsigned u[8]; uint4 q[2]; };
union CFrag { v8f  v; float    f[8]; };

__device__ __forceinline__ unsigned short f2bf(float f) {
  unsigned u = __float_as_uint(f);
  unsigned r = u + 0x7fffu + ((u >> 16) & 1u);   // round-to-nearest-even
  return (unsigned short)(r >> 16);
}
__device__ __forceinline__ unsigned packbf(float a, float b) {
  return (unsigned)f2bf(a) | ((unsigned)f2bf(b) << 16);
}

// ---------------------------------------------------------------------------
// Pack [NF,3072,1024] f32 weights into WMMA B-fragment order:
//   PW[((f*NKT + kt)*NDT + dt)*256 + lane*8 + j] =
//       {bf16(W[f, kt*32 + 8*(lane>>4) + 2j + (j>=4 ? 8 : 0),     dt*16 + (lane&15)]),
//        bf16(W[f, kt*32 + 8*(lane>>4) + 2j + (j>=4 ? 8 : 0) + 1, dt*16 + (lane&15)])}
// so a wave reads one 16x32 B tile as 1KB of contiguous memory (2x b128/lane).
// ---------------------------------------------------------------------------
__global__ __launch_bounds__(256) void pack_w_kernel(const float* __restrict__ W,
                                                     unsigned* __restrict__ PW) {
  __shared__ float slab[32][64];
  const int b   = blockIdx.x;                 // NF*NKT*16 = 6144 blocks
  const int dtg = b & 15;                     // group of 4 d-tiles (64 cols)
  const int kt  = (b >> 4) % NKT;
  const int f   = b / (NKT * 16);
  const int tid = threadIdx.x;

  const float* src = W + ((size_t)f * TD + kt * 32) * D_DIM + dtg * 64;
  #pragma unroll
  for (int i = 0; i < 8; ++i) {               // 2048 floats, rows of 64 contiguous
    int idx = tid + i * 256;
    int row = idx >> 6, col = idx & 63;
    slab[row][col] = src[(size_t)row * D_DIM + col];
  }
  __syncthreads();
  unsigned* dst = PW + (((size_t)f * NKT + kt) * NDT + dtg * 4) * 256;
  #pragma unroll
  for (int i = 0; i < 4; ++i) {               // 1024 dwords out, contiguous
    int o = tid + i * 256;
    int dtl = o >> 8, rem = o & 255;
    int lane = rem >> 3, j = rem & 7;
    int kl = 8 * (lane >> 4) + 2 * j + (j >= 4 ? 8 : 0);
    int dl = dtl * 16 + (lane & 15);
    dst[o] = packbf(slab[kl][dl], slab[kl + 1][dl]);
  }
}

// ---------------------------------------------------------------------------
// Build comb (bf16 pairs): comb[row=n*4+k][c] = x[n,c] | routed[n,k,c-D] |
// delta_table[clip(t-topk+256,0,512), c-2D]
// ---------------------------------------------------------------------------
__global__ __launch_bounds__(256) void build_comb_kernel(
    const float* __restrict__ x, const float* __restrict__ routed,
    const float* __restrict__ dtab, const int* __restrict__ topk,
    unsigned* __restrict__ comb) {
  int u   = blockIdx.x * 256 + threadIdx.x;        // total = NROWS*TD2 = 25165824
  int row = u / TD2;
  int c2  = u % TD2;
  int c   = 2 * c2;
  int n   = row >> 2;
  int t   = n & 1023;                              // n % T
  float f0, f1;
  if (c < D_DIM) {
    const float* p = x + (size_t)n * D_DIM + c;
    f0 = p[0]; f1 = p[1];
  } else if (c < 2 * D_DIM) {
    const float* p = routed + (size_t)row * D_DIM + (c - D_DIM);
    f0 = p[0]; f1 = p[1];
  } else {
    int dl = t - topk[row] + MAXD_;
    dl = dl < 0 ? 0 : (dl > 2 * MAXD_ ? 2 * MAXD_ : dl);
    const float* p = dtab + (size_t)dl * D_DIM + (c - 2 * D_DIM);
    f0 = p[0]; f1 = p[1];
  }
  comb[u] = packbf(f0, f1);
}

// ---------------------------------------------------------------------------
// softmax over sim[n,0..3] -> sw[n,0..3]
// ---------------------------------------------------------------------------
__global__ __launch_bounds__(256) void softmax_sim_kernel(const float* __restrict__ sim,
                                                          float* __restrict__ sw) {
  int n = blockIdx.x * 256 + threadIdx.x;
  if (n >= NTOK) return;
  float4 s = ((const float4*)sim)[n];
  float m = fmaxf(fmaxf(s.x, s.y), fmaxf(s.z, s.w));
  float e0 = expf(s.x - m), e1 = expf(s.y - m), e2 = expf(s.z - m), e3 = expf(s.w - m);
  float inv = 1.0f / (e0 + e1 + e2 + e3);
  ((float4*)sw)[n] = make_float4(e0 * inv, e1 * inv, e2 * inv, e3 * inv);
}

// ---------------------------------------------------------------------------
// fw[n,f] = softmax_f( x[n,:] @ Wsel[:,f] + bsel[f] )   (one wave per token)
// ---------------------------------------------------------------------------
__global__ __launch_bounds__(256) void selector_kernel(const float* __restrict__ x,
                                                       const float* __restrict__ Wsel,
                                                       const float* __restrict__ bsel,
                                                       float* __restrict__ fw) {
  int lane = threadIdx.x & 31;
  int n = (blockIdx.x * 256 + threadIdx.x) >> 5;   // 512 blocks * 8 waves = 4096
  float a0 = 0.f, a1 = 0.f, a2 = 0.f, a3 = 0.f;
  for (int c = lane; c < D_DIM; c += 32) {
    float xv = x[(size_t)n * D_DIM + c];
    float4 w = ((const float4*)Wsel)[c];           // Wsel row c = 4 contiguous floats
    a0 += xv * w.x; a1 += xv * w.y; a2 += xv * w.z; a3 += xv * w.w;
  }
  for (int m = 16; m >= 1; m >>= 1) {
    a0 += __shfl_xor(a0, m, 32); a1 += __shfl_xor(a1, m, 32);
    a2 += __shfl_xor(a2, m, 32); a3 += __shfl_xor(a3, m, 32);
  }
  if (lane == 0) {
    a0 += bsel[0]; a1 += bsel[1]; a2 += bsel[2]; a3 += bsel[3];
    float m = fmaxf(fmaxf(a0, a1), fmaxf(a2, a3));
    float e0 = expf(a0 - m), e1 = expf(a1 - m), e2 = expf(a2 - m), e3 = expf(a3 - m);
    float inv = 1.0f / (e0 + e1 + e2 + e3);
    ((float4*)fw)[n] = make_float4(e0 * inv, e1 * inv, e2 * inv, e3 * inv);
  }
}

__device__ __forceinline__ void load_bfrag(AFrag& b, const unsigned* __restrict__ bp) {
  b.q[0] = *(const uint4*)bp;
  b.q[1] = *(const uint4*)(bp + 4);
}

// ---------------------------------------------------------------------------
// GEMM1 fused: per block = function f, 32 rows (= 8 tokens x 4 K-neighbors)
// x 1024 cols.  h = gelu(comb @ Wl + bl); LayerNorm over D; sw-weighted K-sum.
// 8 waves; wave owns cols [128w,128w+128); 2 row-tiles x 8 col-tiles per wave.
// Double-buffered LDS A tile (1 barrier/stage); all 8 B tiles of a kk-step are
// preloaded and pinned live via sched_barrier so waitcnt pipelines instead of
// stalling (loadcnt<=N progressive waits, not 0).
// ---------------------------------------------------------------------------
__global__ __launch_bounds__(256) void gemm1_kernel(
    const unsigned* __restrict__ comb,     // [NROWS][TD2] bf16-pairs
    const unsigned* __restrict__ pWl,      // B-fragment packed
    const float* __restrict__ bl, const float* __restrict__ lng,
    const float* __restrict__ lnb, const float* __restrict__ sw,
    float* __restrict__ arg) {             // [NF][NTOK][1024]
  const int f    = blockIdx.y;
  const int rb   = blockIdx.x;             // 512 row-blocks of 32 rows
  const int row0 = rb * 32;
  const int tid  = threadIdx.x;
  const int wave = tid >> 5, lane = tid & 31;
  const int lrow = lane & 15, kg = lane >> 4;
  const int dbase = wave * 128;
  const int dt0   = wave * 8;              // first 16-col tile index

  __shared__ unsigned lds[2][32 * 132];    // double-buffered A tile
  __shared__ float redS[32 * 8], redQ[32 * 8];

  CFrag acc[2][8];
  #pragma unroll
  for (int rt = 0; rt < 2; ++rt)
    #pragma unroll
    for (int t = 0; t < 8; ++t)
      #pragma unroll
      for (int i = 0; i < 8; ++i) acc[rt][t].f[i] = 0.f;

  const unsigned* wbase = pWl + (size_t)f * NKT * NDT * 256;
  const int r_fill = tid >> 3, c_fill = (tid & 7) * 16;  // 32 rows x 8 chunks of 16

  // stage s -> buffer p: 32 rows x 128 pair-dwords
  auto stage = [&](int s, int p) {
    const unsigned* src = comb + (size_t)(row0 + r_fill) * TD2 + s * 128 + c_fill;
    unsigned* dst = &lds[p][r_fill * 132 + c_fill];
    #pragma unroll
    for (int j = 0; j < 4; ++j) {
      *(uint4*)(dst + j * 4) = *(const uint4*)(src + j * 4);
    }
  };

  stage(0, 0);
  __syncthreads();
  for (int s = 0; s < NSTAGE; ++s) {
    const int p = s & 1;
    if (s + 1 < NSTAGE) stage(s + 1, p ^ 1);
    #pragma unroll
    for (int kk = 0; kk < 256; kk += 32) {
      const int kt = (s * 256 + kk) >> 5;
      const unsigned* ktile = wbase + ((size_t)kt * NDT + dt0) * 256 + lane * 8;
      AFrag bfr[8];                        // all 8 tiles live simultaneously
      #pragma unroll
      for (int t = 0; t < 8; ++t) load_bfrag(bfr[t], ktile + t * 256);
      AFrag a[2];
      #pragma unroll
      for (int rt = 0; rt < 2; ++rt) {
        const unsigned* ap = &lds[p][(rt * 16 + lrow) * 132 + (kk >> 1) + kg * 4];
        #pragma unroll
        for (int j = 0; j < 4; ++j) { a[rt].u[j] = ap[j]; a[rt].u[4 + j] = ap[8 + j]; }
      }
      __builtin_amdgcn_sched_barrier(0);   // pin all loads above the WMMA stream
      #pragma unroll
      for (int t = 0; t < 8; ++t) {
        acc[0][t].v = __builtin_amdgcn_wmma_f32_16x16x32_bf16(
            false, a[0].v, false, bfr[t].v, (short)0, acc[0][t].v, false, false);
        acc[1][t].v = __builtin_amdgcn_wmma_f32_16x16x32_bf16(
            false, a[1].v, false, bfr[t].v, (short)0, acc[1][t].v, false, false);
      }
    }
    __syncthreads();
  }

  // -------- epilogue: bias + exact GELU, then LayerNorm over D --------
  float p[2][8], q[2][8];
  #pragma unroll
  for (int rt = 0; rt < 2; ++rt)
    #pragma unroll
    for (int i = 0; i < 8; ++i) { p[rt][i] = 0.f; q[rt][i] = 0.f; }
  #pragma unroll
  for (int t = 0; t < 8; ++t) {
    int d = dbase + t * 16 + lrow;
    float bia = bl[f * D_DIM + d];
    #pragma unroll
    for (int rt = 0; rt < 2; ++rt)
      #pragma unroll
      for (int i = 0; i < 8; ++i) {
        float h = acc[rt][t].f[i] + bia;
        float g = 0.5f * h * (1.0f + erff(h * 0.70710678118654752f));
        acc[rt][t].f[i] = g;
        p[rt][i] += g; q[rt][i] += g * g;
      }
  }
  #pragma unroll
  for (int m = 1; m <= 8; m <<= 1)
    #pragma unroll
    for (int rt = 0; rt < 2; ++rt)
      #pragma unroll
      for (int i = 0; i < 8; ++i) {
        p[rt][i] += __shfl_xor(p[rt][i], m, 32);
        q[rt][i] += __shfl_xor(q[rt][i], m, 32);
      }
  if (lrow == 0)
    #pragma unroll
    for (int rt = 0; rt < 2; ++rt)
      #pragma unroll
      for (int i = 0; i < 8; ++i) {
        int row = rt * 16 + i + 8 * kg;
        redS[row * 8 + wave] = p[rt][i];
        redQ[row * 8 + wave] = q[rt][i];
      }
  __syncthreads();
  float mean[2][8], rstd[2][8];
  #pragma unroll
  for (int rt = 0; rt < 2; ++rt)
    #pragma unroll
    for (int i = 0; i < 8; ++i) {
      int row = rt * 16 + i + 8 * kg;
      float s = 0.f, ss = 0.f;
      #pragma unroll
      for (int w = 0; w < 8; ++w) { s += redS[row * 8 + w]; ss += redQ[row * 8 + w]; }
      float mu = s * (1.0f / (float)D_DIM);
      float var = ss * (1.0f / (float)D_DIM) - mu * mu;
      mean[rt][i] = mu;
      rstd[rt][i] = rsqrtf(var + LN_EPS);
    }
  float swv[2][8];
  #pragma unroll
  for (int rt = 0; rt < 2; ++rt)
    #pragma unroll
    for (int i = 0; i < 8; ++i) {
      int row = rt * 16 + i + 8 * kg;
      int tok = rb * 8 + (row >> 2);
      swv[rt][i] = sw[tok * 4 + (row & 3)];
    }
  #pragma unroll
  for (int rt = 0; rt < 2; ++rt) {
    const int nlo = rb * 8 + rt * 4 + kg * 2, nhi = nlo + 1;
    #pragma unroll
    for (int t = 0; t < 8; ++t) {
      int d = dbase + t * 16 + lrow;
      float gv = lng[f * D_DIM + d], bv = lnb[f * D_DIM + d];
      float o0 = 0.f, o1 = 0.f;
      #pragma unroll
      for (int i = 0; i < 4; ++i)
        o0 += swv[rt][i] * ((acc[rt][t].f[i] - mean[rt][i]) * rstd[rt][i] * gv + bv);
      #pragma unroll
      for (int i = 4; i < 8; ++i)
        o1 += swv[rt][i] * ((acc[rt][t].f[i] - mean[rt][i]) * rstd[rt][i] * gv + bv);
      arg[((size_t)f * NTOK + nlo) * D_DIM + d] = o0;
      arg[((size_t)f * NTOK + nhi) * D_DIM + d] = o1;
    }
  }
}

// ---------------------------------------------------------------------------
// GEMM2 fused: per block = 16 tokens x 1024 cols; loop over all 4 functions.
// gin = [x, arg, arg]; gate = sigmoid(gin @ Wg + bg);
// psum += fw*arg, gsum += fw*gate;  uf = psum*gsum.
// ---------------------------------------------------------------------------
__global__ __launch_bounds__(256) void gemm2_kernel(
    const unsigned* __restrict__ comb,     // x (bf16) lives in comb cols [0,1024)
    const unsigned* __restrict__ pWg,
    const float* __restrict__ bg, const float* __restrict__ arg,
    const float* __restrict__ fw, float* __restrict__ uf) {
  const int blk = blockIdx.x;              // 256 blocks of 16 tokens
  const int n0  = blk * 16;
  const int tid = threadIdx.x;
  const int wave = tid >> 5, lane = tid & 31;
  const int lrow = lane & 15, kg = lane >> 4;
  const int dbase = wave * 128;
  const int dt0   = wave * 8;

  __shared__ unsigned lds[2][16 * 132];

  float psum[8][8], gsum[8][8];
  #pragma unroll
  for (int t = 0; t < 8; ++t)
    #pragma unroll
    for (int i = 0; i < 8; ++i) { psum[t][i] = 0.f; gsum[t][i] = 0.f; }

  for (int f = 0; f < NF_FN; ++f) {
    CFrag acc[8];
    #pragma unroll
    for (int t = 0; t < 8; ++t)
      #pragma unroll
      for (int i = 0; i < 8; ++i) acc[t].f[i] = 0.f;

    const unsigned* wbase = pWg + (size_t)f * NKT * NDT * 256;

    // stage s -> buffer p: 16 rows x 128 pair-dwords (mixed x / arg source)
    auto stage = [&](int s, int p) {
      const int ko = s * 256;
      #pragma unroll
      for (int it = 0; it < 8; ++it) {
        int idx = tid + it * 256;
        int r = idx >> 7, c2 = idx & 127;
        int c = ko + 2 * c2;
        unsigned val;
        if (c < D_DIM) {                   // x slice: reuse comb row (n*4) cols [0,D)
          val = comb[(size_t)(n0 + r) * 4 * TD2 + (c >> 1)];
        } else {                           // arg slice (appears twice)
          int j = (c - D_DIM) & 1023;
          const float* ap = arg + ((size_t)f * NTOK + n0 + r) * D_DIM + j;
          val = packbf(ap[0], ap[1]);
        }
        lds[p][r * 132 + c2] = val;
      }
    };

    stage(0, 0);
    __syncthreads();
    for (int s = 0; s < NSTAGE; ++s) {
      const int p = s & 1;
      if (s + 1 < NSTAGE) stage(s + 1, p ^ 1);
      #pragma unroll
      for (int kk = 0; kk < 256; kk += 32) {
        const int kt = (s * 256 + kk) >> 5;
        const unsigned* ktile = wbase + ((size_t)kt * NDT + dt0) * 256 + lane * 8;
        AFrag bfr[8];                      // all 8 tiles live simultaneously
        #pragma unroll
        for (int t = 0; t < 8; ++t) load_bfrag(bfr[t], ktile + t * 256);
        AFrag a;
        const unsigned* ap = &lds[p][lrow * 132 + (kk >> 1) + kg * 4];
        #pragma unroll
        for (int j = 0; j < 4; ++j) { a.u[j] = ap[j]; a.u[4 + j] = ap[8 + j]; }
        __builtin_amdgcn_sched_barrier(0); // pin all loads above the WMMA stream
        #pragma unroll
        for (int t = 0; t < 8; ++t) {
          acc[t].v = __builtin_amdgcn_wmma_f32_16x16x32_bf16(
              false, a.v, false, bfr[t].v, (short)0, acc[t].v, false, false);
        }
      }
      __syncthreads();
    }
    // epilogue for this f
    float fwv[8];
    #pragma unroll
    for (int i = 0; i < 8; ++i) fwv[i] = fw[(n0 + i + 8 * kg) * 4 + f];
    #pragma unroll
    for (int t = 0; t < 8; ++t) {
      int d = dbase + t * 16 + lrow;
      float bgv = bg[f * D_DIM + d];
      #pragma unroll
      for (int i = 0; i < 8; ++i) {
        float hv = acc[t].f[i] + bgv;
        float sg = 1.0f / (1.0f + expf(-hv));
        float av = arg[((size_t)f * NTOK + n0 + i + 8 * kg) * D_DIM + d];
        psum[t][i] += fwv[i] * av;
        gsum[t][i] += fwv[i] * sg;
      }
    }
  }
  #pragma unroll
  for (int t = 0; t < 8; ++t) {
    int d = dbase + t * 16 + lrow;
    #pragma unroll
    for (int i = 0; i < 8; ++i)
      uf[(size_t)(n0 + i + 8 * kg) * D_DIM + d] = psum[t][i] * gsum[t][i];
  }
}

// ---------------------------------------------------------------------------
// VQ: out[n] = codebook[argmin_j ||uf[n]-cb[j]||^2]  (||uf||^2 constant -> drop)
// One block per token; 8 waves x 4 codes each.
// ---------------------------------------------------------------------------
__global__ __launch_bounds__(256) void vq_kernel(const float* __restrict__ uf,
                                                 const float* __restrict__ cb,
                                                 float* __restrict__ out) {
  const int n = blockIdx.x;
  __shared__ float su[D_DIM];
  __shared__ float d2s[NC_];
  __shared__ int encs;
  const int tid = threadIdx.x, wave = tid >> 5, lane = tid & 31;
  for (int i = tid; i < D_DIM; i += 256) su[i] = uf[(size_t)n * D_DIM + i];
  __syncthreads();
  for (int j = wave * 4; j < wave * 4 + 4; ++j) {
    float dot = 0.f, nn = 0.f;
    for (int e = lane; e < D_DIM; e += 32) {
      float c = cb[(size_t)j * D_DIM + e];
      dot += su[e] * c; nn += c * c;
    }
    for (int m = 16; m >= 1; m >>= 1) {
      dot += __shfl_xor(dot, m, 32);
      nn  += __shfl_xor(nn,  m, 32);
    }
    if (lane == 0) d2s[j] = nn - 2.0f * dot;
  }
  __syncthreads();
  if (tid == 0) {
    int best = 0; float bv = d2s[0];
    for (int j = 1; j < NC_; ++j) if (d2s[j] < bv) { bv = d2s[j]; best = j; }
    encs = best;
  }
  __syncthreads();
  for (int i = tid; i < D_DIM; i += 256)
    out[(size_t)n * D_DIM + i] = cb[(size_t)encs * D_DIM + i];
}

// ---------------------------------------------------------------------------
extern "C" void kernel_launch(void* const* d_in, const int* in_sizes, int n_in,
                              void* d_out, int out_size, void* d_ws, size_t ws_size,
                              hipStream_t stream) {
  const float* x      = (const float*)d_in[0];
  const float* routed = (const float*)d_in[1];
  const float* sim    = (const float*)d_in[2];
  const float* Wl     = (const float*)d_in[3];
  const float* bl     = (const float*)d_in[4];
  const float* lng    = (const float*)d_in[5];
  const float* lnb    = (const float*)d_in[6];
  const float* Wg     = (const float*)d_in[7];
  const float* bg     = (const float*)d_in[8];
  const float* Wsel   = (const float*)d_in[9];
  const float* bsel   = (const float*)d_in[10];
  const float* dtab   = (const float*)d_in[11];
  const float* cb     = (const float*)d_in[12];
  const int*   topk   = (const int*)d_in[13];
  float* out = (float*)d_out;

  char* w = (char*)d_ws;
  unsigned* comb = (unsigned*)w;  w += (size_t)NROWS * TD2 * 4;         // 100.7 MB
  unsigned* pWl  = (unsigned*)w;  w += (size_t)NF_FN * TD2 * D_DIM * 4; //  25.2 MB
  unsigned* pWg  = (unsigned*)w;  w += (size_t)NF_FN * TD2 * D_DIM * 4; //  25.2 MB
  float*    argv = (float*)w;     w += (size_t)NF_FN * NTOK * D_DIM * 4;//  67.1 MB
  float*    sw   = (float*)w;     w += (size_t)NTOK * 4 * 4;
  float*    fw   = (float*)w;     w += (size_t)NTOK * 4 * 4;
  float*    uf   = (float*)w;     w += (size_t)NTOK * D_DIM * 4;        //  16.8 MB

  pack_w_kernel<<<NF_FN * NKT * 16, 256, 0, stream>>>(Wl, pWl);
  pack_w_kernel<<<NF_FN * NKT * 16, 256, 0, stream>>>(Wg, pWg);
  build_comb_kernel<<<98304, 256, 0, stream>>>(x, routed, dtab, topk, comb);
  softmax_sim_kernel<<<16, 256, 0, stream>>>(sim, sw);
  selector_kernel<<<512, 256, 0, stream>>>(x, Wsel, bsel, fw);
  gemm1_kernel<<<dim3(512, NF_FN), 256, 0, stream>>>(comb, pWl, bl, lng, lnb, sw, argv);
  gemm2_kernel<<<256, 256, 0, stream>>>(comb, pWg, bg, argv, fw, uf);
  vq_kernel<<<NTOK, 256, 0, stream>>>(uf, cb, out);
}
